// MambaEEGModel_71829033059129
// MI455X (gfx1250) — compile-verified, matrix operand order checked by reading, not run on hardware
//
#include <hip/hip_runtime.h>
#include <hip/hip_bf16.h>
#include <math.h>

// ---------------- model constants ----------------
#define LSEQ    2048
#define BATCH   8
#define MTOK    (BATCH*LSEQ)      // 16384 tokens
#define DMODEL  256
#define DINNER  512
#define DSTATE  16
#define NHEADS  16
#define HEADDIM 32
#define CONVDIM 544               // DINNER + 2*DSTATE
#define DINPROJ 1072              // 2*DINNER + 2*DSTATE + NHEADS
#define ZXLD    1088              // DINPROJ padded to 68 WMMA tiles (17 waves x 4)
#define NLAYERS 4
#define NFILT   64
#define NCH     64
#define EPSLN   1e-5f

// ---------------- WMMA vector types ----------------
typedef __attribute__((ext_vector_type(16))) __bf16          v16bf;
typedef __attribute__((ext_vector_type(16))) unsigned short  v16u16;
typedef __attribute__((ext_vector_type(8)))  unsigned short  v8u16;
typedef __attribute__((ext_vector_type(8)))  float           v8f;

// optional gfx1250 async global->LDS path (compile-safe fallback)
#if defined(__has_builtin)
#if __has_builtin(__builtin_amdgcn_global_load_async_to_lds_b128) && \
    __has_builtin(__builtin_amdgcn_s_wait_asynccnt)
#define USE_ASYNC_LDS 1
#endif
#endif
#ifdef USE_ASYNC_LDS
// builtin signature (from hipcc diagnostic): param1 = global int4*, param2 = shared int4*
typedef int v4i_g __attribute__((vector_size(16)));
typedef __attribute__((address_space(1))) v4i_g* gv4i_p;   // global (AS1)
typedef __attribute__((address_space(3))) v4i_g* lv4i_p;   // LDS (AS3)
#endif

static __device__ inline v16bf as_v16bf(v16u16 x) {
    union { v16u16 u; v16bf b; } c; c.u = x; return c.b;
}

// load one 16x32 fragment row slice: 8 halves at p, 8 at p+16
static __device__ inline v16bf frag16(const unsigned short* p) {
    v8u16 lo = *(const v8u16*)p;
    v8u16 hi = *(const v8u16*)(p + 16);
    v16u16 v;
    #pragma unroll
    for (int i = 0; i < 8; ++i) { v[i] = lo[i]; v[i + 8] = hi[i]; }
    return as_v16bf(v);
}

static __device__ inline unsigned short f32_to_bf16(float f) {
    unsigned int u = __float_as_uint(f);
    u += 0x7FFFu + ((u >> 16) & 1u);           // round-to-nearest-even
    return (unsigned short)(u >> 16);
}

static __device__ inline float block_reduce_sum(float v, float* sbuf) {
    int tid = threadIdx.x;
    sbuf[tid] = v; __syncthreads();
    for (int s = blockDim.x >> 1; s > 0; s >>= 1) {
        if (tid < s) sbuf[tid] += sbuf[tid + s];
        __syncthreads();
    }
    float r = sbuf[0]; __syncthreads();
    return r;
}

static __device__ inline float silu_f(float x)  { return x / (1.f + __expf(-x)); }
static __device__ inline float sftpls(float x)  { return (x > 20.f) ? x : log1pf(__expf(x)); }

// ====================================================================
// 0) weight cast fp32 -> bf16; in_proj padded 1072 -> 1088 rows (zeros)
// ====================================================================
__global__ void k_castw(const float* __restrict__ ip, const float* __restrict__ op,
                        unsigned short* __restrict__ ipb, unsigned short* __restrict__ opb) {
    const int n_ip = NLAYERS * ZXLD * DMODEL;        // padded
    const int n_op = NLAYERS * DMODEL * DINNER;
    for (int i = blockIdx.x * blockDim.x + threadIdx.x; i < n_ip + n_op;
         i += gridDim.x * blockDim.x) {
        if (i < n_ip) {
            int k = i % DMODEL;
            int n = (i / DMODEL) % ZXLD;
            int l = i / (DMODEL * ZXLD);
            ipb[i] = (n < DINPROJ)
                   ? f32_to_bf16(ip[((size_t)l * DINPROJ + n) * DMODEL + k])
                   : (unsigned short)0;
        } else {
            opb[i - n_ip] = f32_to_bf16(op[i - n_ip]);
        }
    }
}

// ====================================================================
// 0b) channel-embed mean + negA = -exp(A_log) (single block)
// ====================================================================
__global__ void k_prep_small(const float* __restrict__ ce, const float* __restrict__ Alog,
                             float* __restrict__ cemean, float* __restrict__ negA) {
    int d = threadIdx.x;   // 0..255
    float s = 0.f;
    for (int r = 0; r < NCH; ++r) s += ce[r * DMODEL + d];
    cemean[d] = s * (1.f / NCH);
    if (d < NLAYERS * NHEADS) negA[d] = -__expf(Alog[d]);
}

// ====================================================================
// 1) front-end conv: xs[b,f,l] = conv1d(x, sconv_w, pad=1) + sconv_b
// ====================================================================
__global__ void k_sconv(const float* __restrict__ x, const float* __restrict__ w,
                        const float* __restrict__ bias, float* __restrict__ xs) {
    int b = blockIdx.z, f = blockIdx.y;
    int l = blockIdx.x * 128 + threadIdx.x;
    __shared__ float wsm[NCH * 3];
    __shared__ float bsh;
    for (int i = threadIdx.x; i < NCH * 3; i += 128) wsm[i] = w[f * NCH * 3 + i];
    if (threadIdx.x == 0) bsh = bias[f];
    __syncthreads();
    float acc = bsh;
    const float* xb = x + (size_t)b * NCH * LSEQ;
    for (int c = 0; c < NCH; ++c) {
        const float* xc = xb + (size_t)c * LSEQ;
        float x0 = (l - 1 >= 0)   ? xc[l - 1] : 0.f;
        float x1 = xc[l];
        float x2 = (l + 1 < LSEQ) ? xc[l + 1] : 0.f;
        acc += x0 * wsm[c * 3] + x1 * wsm[c * 3 + 1] + x2 * wsm[c * 3 + 2];
    }
    xs[((size_t)b * NFILT + f) * LSEQ + l] = acc;
}

// ====================================================================
// 2) mixer + enc layernorm: h[b,l,d], also bf16 copy for WMMA A
// ====================================================================
__global__ void k_mixer_ln(const float* __restrict__ xs, const float* __restrict__ mw,
                           const float* __restrict__ mb, const float* __restrict__ cemean,
                           const float* __restrict__ g, const float* __restrict__ bta,
                           float* __restrict__ h, unsigned short* __restrict__ hbf) {
    int tok = blockIdx.x, b = tok >> 11, l = tok & (LSEQ - 1);
    int d = threadIdx.x;
    __shared__ float xsm[NFILT];
    __shared__ float red[256];
    if (d < NFILT) xsm[d] = xs[((size_t)b * NFILT + d) * LSEQ + l];
    __syncthreads();
    float acc = mb[d] + cemean[d];
    #pragma unroll 8
    for (int f = 0; f < NFILT; ++f) acc += xsm[f] * mw[d * NFILT + f];
    float mu  = block_reduce_sum(acc, red) * (1.f / DMODEL);
    float dv  = acc - mu;
    float var = block_reduce_sum(dv * dv, red) * (1.f / DMODEL);
    float o   = dv * rsqrtf(var + EPSLN) * g[d] + bta[d];
    h  [(size_t)tok * DMODEL + d] = o;
    hbf[(size_t)tok * DMODEL + d] = f32_to_bf16(o);
}

// ====================================================================
// 3) bf16 WMMA GEMM:  C[M,N] = A[M,K] * W[N,K]^T   (fp32 accumulate)
//    block = 128 threads (4 waves); A tile staged in LDS (async if
//    available); each active wave owns exactly 4 full N-tiles, so the
//    K-loop is branch-free and EXEC stays all-ones around every WMMA.
//    Ntiles MUST be a multiple of 4 (padded by caller).
// ====================================================================
__global__ __launch_bounds__(128) void k_gemm_bf16(
        const unsigned short* __restrict__ A,   // M x K (bf16 bits)
        const unsigned short* __restrict__ W,   // N x K (bf16 bits)
        float* __restrict__ C, int K, int Ntiles, int ldC) {
    const int lane  = threadIdx.x & 31;
    const int wave  = threadIdx.x >> 5;
    const int tileM = blockIdx.y;
    const int tn0   = (blockIdx.x * 4 + wave) * 4;

    // ---- stage the 16 x K A tile into LDS (whole block cooperates) ----
    __shared__ __align__(16) unsigned short lA[17 * 512];   // 17 rows: tail over-read pad
    {
        const int cpr    = K >> 3;            // 16-byte chunks per row
        const int chunks = 16 * cpr;
        for (int i = threadIdx.x; i < chunks; i += 128) {
            int row = i / cpr, cc = i - row * cpr;
            const unsigned short* g = A + ((size_t)(tileM * 16 + row)) * K + cc * 8;
            unsigned short*       l = &lA[row * K + cc * 8];
#ifdef USE_ASYNC_LDS
            __builtin_amdgcn_global_load_async_to_lds_b128(
                (gv4i_p)g, (lv4i_p)l, 0, 0);
#else
            *(v8u16*)l = *(const v8u16*)g;
#endif
        }
#ifdef USE_ASYNC_LDS
        __builtin_amdgcn_s_wait_asynccnt(0);
#endif
        __syncthreads();
    }
    if (tn0 >= Ntiles) return;                 // uniform per wave, after barrier

    const int rlo  = lane & 15;
    const int koff = (lane >> 4) << 3;         // 0 or 8 (A/B fragment layout)
    const unsigned short* la    = &lA[rlo * K + koff];
    const unsigned short* wrow0 = W + ((size_t)((tn0 + 0) * 16 + rlo)) * K + koff;
    const unsigned short* wrow1 = W + ((size_t)((tn0 + 1) * 16 + rlo)) * K + koff;
    const unsigned short* wrow2 = W + ((size_t)((tn0 + 2) * 16 + rlo)) * K + koff;
    const unsigned short* wrow3 = W + ((size_t)((tn0 + 3) * 16 + rlo)) * K + koff;

    v8f acc0 = {}, acc1 = {}, acc2 = {}, acc3 = {};

    // ---- software-pipelined K loop (2-stage; branch-free tail via
    //      harmless 64B over-read that stays inside ws / padded LDS) ----
    v16bf aC = frag16(la);
    v16bf b0 = frag16(wrow0), b1 = frag16(wrow1);
    v16bf b2 = frag16(wrow2), b3 = frag16(wrow3);

    for (int k0 = 0; k0 < K; k0 += 32) {
        const int kn = k0 + 32;
        __builtin_prefetch(wrow0 + kn + 32, 0, 1);       // global_prefetch_b8
        v16bf aN  = frag16(la + kn);
        v16bf b0n = frag16(wrow0 + kn);
        v16bf b1n = frag16(wrow1 + kn);
        v16bf b2n = frag16(wrow2 + kn);
        v16bf b3n = frag16(wrow3 + kn);
        acc0 = __builtin_amdgcn_wmma_f32_16x16x32_bf16(false, aC, false, b0, (short)0, acc0, false, false);
        acc1 = __builtin_amdgcn_wmma_f32_16x16x32_bf16(false, aC, false, b1, (short)0, acc1, false, false);
        acc2 = __builtin_amdgcn_wmma_f32_16x16x32_bf16(false, aC, false, b2, (short)0, acc2, false, false);
        acc3 = __builtin_amdgcn_wmma_f32_16x16x32_bf16(false, aC, false, b3, (short)0, acc3, false, false);
        aC = aN; b0 = b0n; b1 = b1n; b2 = b2n; b3 = b3n;
    }

    // ---- store (f32 C/D layout: elem r -> row rbase+r, col = lane&15) ----
    const int rbase = tileM * 16 + ((lane >> 4) ? 8 : 0);
    {
        const int cn0 = (tn0 + 0) * 16 + rlo;
        const int cn1 = (tn0 + 1) * 16 + rlo;
        const int cn2 = (tn0 + 2) * 16 + rlo;
        const int cn3 = (tn0 + 3) * 16 + rlo;
        #pragma unroll
        for (int r = 0; r < 8; ++r) {
            float* crow = C + (size_t)(rbase + r) * ldC;
            crow[cn0] = acc0[r];
            crow[cn1] = acc1[r];
            crow[cn2] = acc2[r];
            crow[cn3] = acc3[r];
        }
    }
}

// ====================================================================
// 4) depthwise causal conv (D_CONV=4) + SiLU on xBC; dt softplus + dA
// ====================================================================
__global__ void k_conv_silu(const float* __restrict__ zx, const float* __restrict__ cw,
                            const float* __restrict__ cb, const float* __restrict__ dtb,
                            const float* __restrict__ negA, float* __restrict__ xact,
                            float* __restrict__ dts, float* __restrict__ dAb) {
    int tok = blockIdx.x;
    int l   = tok & (LSEQ - 1);
    int c   = threadIdx.x;                               // 0..543
    if (c < CONVDIM) {
        float acc = cb[c];
        #pragma unroll
        for (int t = 0; t < 4; ++t) {
            int li = l - 3 + t;
            if (li >= 0)
                acc += zx[(size_t)(tok - 3 + t) * ZXLD + DINNER + c] * cw[c * 4 + t];
        }
        xact[(size_t)tok * CONVDIM + c] = silu_f(acc);
    }
    if (c < NHEADS) {
        float xv = zx[(size_t)tok * ZXLD + (DINNER + CONVDIM) + c] + dtb[c];
        float dt = sftpls(xv);
        dts[(size_t)tok * NHEADS + c] = dt;
        dAb[(size_t)tok * NHEADS + c] = __expf(dt * negA[c]);
    }
}

// ====================================================================
// 5) SSM scan: one block per (batch, head); thread = (p, n) state elem
// ====================================================================
__global__ __launch_bounds__(512) void k_scan(
        const float* __restrict__ xact, const float* __restrict__ dts,
        const float* __restrict__ dAb, const float* __restrict__ Ds,
        float* __restrict__ y) {
    int bh = blockIdx.x, b = bh >> 4, hd = bh & 15;
    int tid = threadIdx.x, p = tid >> 4, n = tid & 15;
    float s = 0.f;
    float dskip = Ds[hd];
    size_t tok = (size_t)b * LSEQ;
    for (int l = 0; l < LSEQ; ++l, ++tok) {
        const float* xr = xact + tok * CONVDIM;
        float xp = xr[hd * HEADDIM + p];
        float Bn = xr[DINNER + n];
        float Cn = xr[DINNER + DSTATE + n];
        float dt = dts[tok * NHEADS + hd];
        float dA = dAb[tok * NHEADS + hd];
        s = s * dA + dt * xp * Bn;
        float part = s * Cn;                       // reduce over n (16 lanes)
        part += __shfl_xor(part, 1, 32);
        part += __shfl_xor(part, 2, 32);
        part += __shfl_xor(part, 4, 32);
        part += __shfl_xor(part, 8, 32);
        if (n == 0)
            y[tok * DINNER + hd * HEADDIM + p] = part + dskip * xp;
    }
}

// ====================================================================
// 6) gate (silu(z)) + RMSNorm + bf16 cast for out_proj GEMM A
// ====================================================================
__global__ void k_gate_rms(const float* __restrict__ yraw, const float* __restrict__ zx,
                           const float* __restrict__ nw, unsigned short* __restrict__ ybf) {
    int tok = blockIdx.x, c = threadIdx.x;
    __shared__ float red[256];
    float z0 = zx[(size_t)tok * ZXLD + c];
    float z1 = zx[(size_t)tok * ZXLD + c + 256];
    float v0 = yraw[(size_t)tok * DINNER + c]       * silu_f(z0);
    float v1 = yraw[(size_t)tok * DINNER + c + 256] * silu_f(z1);
    float ms = block_reduce_sum(v0 * v0 + v1 * v1, red) * (1.f / DINNER);
    float r  = rsqrtf(ms + EPSLN);
    ybf[(size_t)tok * DINNER + c]       = f32_to_bf16(v0 * r * nw[c]);
    ybf[(size_t)tok * DINNER + c + 256] = f32_to_bf16(v1 * r * nw[c + 256]);
}

// ====================================================================
// 7) residual add + layernorm; refresh fp32 h and bf16 hbf
// ====================================================================
__global__ void k_residual_ln(const float* __restrict__ om, float* __restrict__ h,
                              unsigned short* __restrict__ hbf, const float* __restrict__ g,
                              const float* __restrict__ bta) {
    int tok = blockIdx.x, d = threadIdx.x;
    __shared__ float red[256];
    float v   = om[(size_t)tok * DMODEL + d] + h[(size_t)tok * DMODEL + d];
    float mu  = block_reduce_sum(v, red) * (1.f / DMODEL);
    float dv  = v - mu;
    float var = block_reduce_sum(dv * dv, red) * (1.f / DMODEL);
    float o   = dv * rsqrtf(var + EPSLN) * g[d] + bta[d];
    h  [(size_t)tok * DMODEL + d] = o;
    hbf[(size_t)tok * DMODEL + d] = f32_to_bf16(o);
}

// ====================================================================
// 8) head: time-mean -> GEMV -> GELU -> GEMV
// ====================================================================
__global__ void k_head(const float* __restrict__ h, const float* __restrict__ w1,
                       const float* __restrict__ b1, const float* __restrict__ w2,
                       const float* __restrict__ b2, float* __restrict__ out) {
    int b = blockIdx.x, d = threadIdx.x;
    __shared__ float p[DMODEL];
    __shared__ float hid[DMODEL / 2];
    __shared__ float red[256];
    float acc = 0.f;
    const float* hb = h + (size_t)b * LSEQ * DMODEL;
    for (int l = 0; l < LSEQ; ++l) acc += hb[(size_t)l * DMODEL + d];
    p[d] = acc * (1.f / LSEQ);
    __syncthreads();
    if (d < DMODEL / 2) {
        float a = b1[d];
        for (int k = 0; k < DMODEL; ++k) a += p[k] * w1[d * DMODEL + k];
        hid[d] = 0.5f * a * (1.f + erff(a * 0.70710678118f));   // exact GELU
    }
    __syncthreads();
    float t = (d < DMODEL / 2) ? hid[d] * w2[d] : 0.f;
    float s = block_reduce_sum(t, red);
    if (d == 0) out[b] = s + b2[0];
}

// ====================================================================
// launch
// ====================================================================
extern "C" void kernel_launch(void* const* d_in, const int* in_sizes, int n_in,
                              void* d_out, int out_size, void* d_ws, size_t ws_size,
                              hipStream_t stream) {
    (void)in_sizes; (void)n_in; (void)out_size; (void)ws_size;
    const float* x         = (const float*)d_in[0];
    const float* chanemb   = (const float*)d_in[1];
    const float* sconv_w   = (const float*)d_in[2];
    const float* sconv_b   = (const float*)d_in[3];
    const float* mixer_w   = (const float*)d_in[4];
    const float* mixer_b   = (const float*)d_in[5];
    const float* enc_ln_g  = (const float*)d_in[6];
    const float* enc_ln_b  = (const float*)d_in[7];
    const float* in_proj_w = (const float*)d_in[8];
    const float* conv_w    = (const float*)d_in[9];
    const float* conv_b    = (const float*)d_in[10];
    const float* dt_bias   = (const float*)d_in[11];
    const float* A_log     = (const float*)d_in[12];
    const float* D_skip    = (const float*)d_in[13];
    const float* norm_w    = (const float*)d_in[14];
    const float* out_projw = (const float*)d_in[15];
    const float* ln_g      = (const float*)d_in[16];
    const float* ln_b      = (const float*)d_in[17];
    const float* head_w1   = (const float*)d_in[18];
    const float* head_b1   = (const float*)d_in[19];
    const float* head_w2   = (const float*)d_in[20];
    const float* head_b2   = (const float*)d_in[21];
    float* outp = (float*)d_out;

    // ---- workspace carving (everything L2-resident; ~190 MB total) ----
    size_t off = 0;
    auto take = [&](size_t bytes) -> char* {
        char* r = (char*)d_ws + off;
        off += (bytes + 255) & ~(size_t)255;
        return r;
    };
    float*          h    = (float*)         take((size_t)MTOK * DMODEL * 4);
    unsigned short* hbf  = (unsigned short*)take((size_t)MTOK * DMODEL * 2);
    float*          zx   = (float*)         take((size_t)MTOK * ZXLD * 4);   // also xs (front-end)
    float*          xact = (float*)         take((size_t)MTOK * CONVDIM * 4);
    float*          ybuf = (float*)         take((size_t)MTOK * DINNER * 4); // y_raw; reused as out_mm
    unsigned short* ybf  = (unsigned short*)take((size_t)MTOK * DINNER * 2);
    float*          dts  = (float*)         take((size_t)MTOK * NHEADS * 4);
    float*          dAb  = (float*)         take((size_t)MTOK * NHEADS * 4);
    unsigned short* wipb = (unsigned short*)take((size_t)NLAYERS * ZXLD * DMODEL * 2);
    unsigned short* wopb = (unsigned short*)take((size_t)NLAYERS * DMODEL * DINNER * 2);
    float*          cem  = (float*)         take(DMODEL * 4);
    float*          negA = (float*)         take(NLAYERS * NHEADS * 4);
    float*          xs   = zx;                 // front-end scratch aliases zx
    float*          omm  = ybuf;               // out_proj result aliases y_raw

    // ---- prep ----
    k_prep_small<<<1, 256, 0, stream>>>(chanemb, A_log, cem, negA);
    k_castw<<<2048, 256, 0, stream>>>(in_proj_w, out_projw, wipb, wopb);

    // ---- front-end ----
    k_sconv<<<dim3(LSEQ / 128, NFILT, BATCH), 128, 0, stream>>>(x, sconv_w, sconv_b, xs);
    k_mixer_ln<<<MTOK, 256, 0, stream>>>(xs, mixer_w, mixer_b, cem, enc_ln_g, enc_ln_b, h, hbf);

    // ---- mamba layers ----
    const int nt1 = ZXLD / 16;      // 68 N-tiles (17 waves x 4, no partial waves)
    for (int layer = 0; layer < NLAYERS; ++layer) {
        // zxbcdt = h @ in_proj^T  : [MTOK,256] x [1088,256]^T (padded)
        k_gemm_bf16<<<dim3((nt1 / 4 + 3) / 4, MTOK / 16), 128, 0, stream>>>(
            hbf, wipb + (size_t)layer * ZXLD * DMODEL, zx, DMODEL, nt1, ZXLD);
        // depthwise conv + silu + dt/dA
        k_conv_silu<<<MTOK, CONVDIM, 0, stream>>>(
            zx, conv_w + (size_t)layer * CONVDIM * 4, conv_b + (size_t)layer * CONVDIM,
            dt_bias + layer * NHEADS, negA + layer * NHEADS, xact, dts, dAb);
        // sequential SSM scan
        k_scan<<<BATCH * NHEADS, 512, 0, stream>>>(xact, dts, dAb, D_skip + layer * NHEADS, ybuf);
        // gate + rmsnorm -> bf16
        k_gate_rms<<<MTOK, 256, 0, stream>>>(ybuf, zx, norm_w + (size_t)layer * DINNER, ybf);
        // out = y @ out_proj^T : [MTOK,512] x [256,512]^T  (writes over y_raw)
        k_gemm_bf16<<<dim3(1, MTOK / 16), 128, 0, stream>>>(
            ybf, wopb + (size_t)layer * DMODEL * DINNER, omm, DINNER, DMODEL / 16, DMODEL);
        // residual + layernorm
        k_residual_ln<<<MTOK, 256, 0, stream>>>(
            omm, h, hbf, ln_g + (size_t)layer * DMODEL, ln_b + (size_t)layer * DMODEL);
    }

    // ---- head ----
    k_head<<<BATCH, 256, 0, stream>>>(h, head_w1, head_b1, head_w2, head_b2, outp);
}
